// LSTM_Mod_21388937134510
// MI455X (gfx1250) — matrix operand order repaired and checked by reading, not compiled, not run on hardware
//
#include <hip/hip_runtime.h>
#include <stdint.h>

#define HID 1024
#define SEQ 8192
#define NWG 64        // each workgroup owns 16 hidden units (64 rows of W_hh)
#define TPB 256       // 8 wave32 waves
#define WSTRIDE 1032  // padded LDS row stride (bf16 elems) -> bank-conflict free

typedef __attribute__((ext_vector_type(16))) __bf16 bf16x16;
typedef __attribute__((ext_vector_type(8)))  __bf16 bf16x8;
typedef __attribute__((ext_vector_type(8)))  float  f32x8;
typedef __attribute__((ext_vector_type(4)))  unsigned int u32x4;
typedef __attribute__((ext_vector_type(8)))  int i32x8;
typedef __attribute__((ext_vector_type(4)))  int i32x4;

// dynamic-LDS layout (bytes)
#define OFF_W     0u                 // 64 * 1032 * 2 = 132096 B  (bf16 weight shard)
#define OFF_STAGE 132096u            // 4 * 1024 * 4  = 16384 B   (TDM f32 staging)
#define OFF_H     148480u            // 1024 * 2      = 2048 B    (bf16 h broadcast)
#define OFF_Z     150528u            // 64 * 4                    (gate preacts)
#define OFF_WIH   150784u            // 64 * 4
#define OFF_BIAS  151040u            // 64 * 4
#define OFF_WLIN  151296u            // 16 * 4
#define OFF_C     151360u            // 16 * 4
#define SMEM_BYTES 151424u

#if defined(__has_builtin)
#if __has_builtin(__builtin_amdgcn_tensor_load_to_lds)
#define USE_TDM 1
#endif
#endif

__device__ __forceinline__ float sigf(float x) { return 1.0f / (1.0f + __expf(-x)); }
__device__ __forceinline__ unsigned short f2bf(float x) {
  return (unsigned short)(__float_as_uint(x) >> 16);
}

__global__ void __launch_bounds__(TPB, 1)
lstm_init_kernel(float* __restrict__ out, const float* __restrict__ b_lin,
                 unsigned* __restrict__ cnt, unsigned short* __restrict__ hbuf) {
  const int i = blockIdx.x * blockDim.x + threadIdx.x;
  if (i < SEQ) out[i] = b_lin[0];
  if (i < 2 * HID) hbuf[i] = 0;           // h0 = 0 (both double buffers)
  if (i == 0) *cnt = 0u;                  // global-barrier counter
}

__global__ void __launch_bounds__(TPB, 1)
lstm_persistent_kernel(const float* __restrict__ sentence,
                       const float* __restrict__ W_ih,
                       const float* __restrict__ W_hh,
                       const float* __restrict__ b_ih,
                       const float* __restrict__ b_hh,
                       const float* __restrict__ W_lin,
                       float* __restrict__ out,
                       unsigned* __restrict__ cnt,
                       unsigned short* __restrict__ hbuf) {
  extern __shared__ unsigned char smem[];
  unsigned short* __restrict__ Wlds   = (unsigned short*)(smem + OFF_W);
  float*          __restrict__ stage  = (float*)(smem + OFF_STAGE);
  unsigned short* __restrict__ h_lds  = (unsigned short*)(smem + OFF_H);
  float*          __restrict__ z_lds  = (float*)(smem + OFF_Z);
  float*          __restrict__ wih_l  = (float*)(smem + OFF_WIH);
  float*          __restrict__ bias_l = (float*)(smem + OFF_BIAS);
  float*          __restrict__ wlin_l = (float*)(smem + OFF_WLIN);
  float*          __restrict__ c_l    = (float*)(smem + OFF_C);

  const int tid  = threadIdx.x;
  const int wgid = blockIdx.x;
  const int wid  = tid >> 5;
  const int lane = tid & 31;

  // ---- setup: per-WG constants ----------------------------------------
  if (tid < 64) {
    const int g = tid >> 4, r = tid & 15;
    const int row = g * HID + wgid * 16 + r;     // row of the 4H x H system
    wih_l[tid]  = W_ih[row];
    bias_l[tid] = b_ih[row] + b_hh[row];
  }
  if (tid < 16) {
    c_l[tid]    = 0.0f;
    wlin_l[tid] = W_lin[wgid * 16 + tid];
  }

  // ---- setup: stage W_hh shard into LDS as bf16 (TDM if available) ----
#ifdef USE_TDM
  for (int ch = 0; ch < 16; ++ch) {              // 16 chunks of 4 rows
    const int g = ch >> 2, r0 = (ch & 3) * 4;
    const unsigned grow = (unsigned)g * (unsigned)HID + (unsigned)wgid * 16u + (unsigned)r0;
    if (tid < 32) {                              // wave 0 issues the TDM op
      const uint64_t ga = (uint64_t)(uintptr_t)W_hh + (uint64_t)grow * (uint64_t)(HID * 4);
      const unsigned lds_off = (unsigned)(uintptr_t)(void*)stage;
      u32x4 g0 = { 1u,                                     // count = 1 valid D#
                   lds_off,                                // lds_addr
                   (unsigned)ga,                           // global_addr[31:0]
                   (unsigned)(ga >> 32) | (2u << 30) };    // global_addr[56:32] | type=2
      i32x8 g1 = { (int)(2u << 16),          // data_size = 4B
                   (int)(1024u << 16),       // tensor_dim0[15:0] = 1024
                   (int)(4096u << 16),       // tensor_dim1[15:0] = 4096
                   (int)(1024u << 16),       // tile_dim0 = 1024
                   4,                        // tile_dim1 = 4 rows
                   1024,                     // tensor_dim0_stride = 1024
                   0, 0 };
      i32x4 g2 = { 0, 0, 0, 0 };
      i32x4 g3 = { 0, 0, 0, 0 };
      i32x8 g4 = { 0, 0, 0, 0, 0, 0, 0, 0 };
      __builtin_amdgcn_tensor_load_to_lds(g0, g1, g2, g3, g4, 0);
      __builtin_amdgcn_s_wait_tensorcnt(0);
    }
    __syncthreads();
    for (int i = tid; i < 4 * HID; i += TPB) {   // f32 -> bf16 into padded shard
      const int rr = i >> 10, k = i & (HID - 1);
      Wlds[(unsigned)(g * 16 + r0 + rr) * WSTRIDE + k] = f2bf(stage[i]);
    }
    __syncthreads();
  }
#else
  for (int i = tid; i < 64 * HID; i += TPB) {
    const int lr = i >> 10, k = i & (HID - 1);
    const int g = lr >> 4, r = lr & 15;
    const float w = W_hh[((size_t)(g * HID + wgid * 16 + r) << 10) + k];
    Wlds[(unsigned)lr * WSTRIDE + k] = f2bf(w);
  }
  __syncthreads();
#endif

  // wave -> (gate tile, k half) assignment for the WMMA matvec
  const int gw   = wid >> 1;                  // gate 0..3
  const int kh   = (wid & 1) * 512;           // k half: [0,512) or [512,1024)
  const int m    = lane & 15;                 // A-tile row within lane group
  const int loff = (lane < 16) ? 0 : 8;       // 16-bit A layout K sub-offset
  const unsigned short* __restrict__ wrow = Wlds + (unsigned)(gw * 16 + m) * WSTRIDE;

  // ---- sequential scan over timesteps ---------------------------------
  for (int t = 0; t < SEQ; ++t) {
    const unsigned short* __restrict__ h_src = hbuf + (size_t)(t & 1) * HID;
    unsigned short*       __restrict__ h_dst = hbuf + (size_t)((t + 1) & 1) * HID;

    if (tid < 64) z_lds[tid] = 0.0f;
    // broadcast h (1024 bf16 = 2 KB) into LDS, 8 B per thread
    ((unsigned long long*)h_lds)[tid] = ((const unsigned long long*)h_src)[tid];
    if (tid == 0) __builtin_prefetch(sentence + t + 32, 0, 1);
    __syncthreads();

    // 16 chained bf16 WMMAs: this wave's 16x512 slice of W_hh times h
    f32x8 acc = {};
#pragma unroll
    for (int kt = 0; kt < 16; ++kt) {
      const int k0 = kh + kt * 32;
      bf16x8 alo = *(const bf16x8*)(wrow + k0 + loff);
      bf16x8 ahi = *(const bf16x8*)(wrow + k0 + 16 + loff);
      bf16x16 A = __builtin_shufflevector(alo, ahi,
          0, 1, 2, 3, 4, 5, 6, 7, 8, 9, 10, 11, 12, 13, 14, 15);
      const int hb = k0 + ((lane < 16) ? 0 : 8);
      bf16x8 blo = *(const bf16x8*)(h_lds + hb);
      bf16x8 bhi = *(const bf16x8*)(h_lds + hb + 16);
      bf16x16 B = __builtin_shufflevector(blo, bhi,
          0, 1, 2, 3, 4, 5, 6, 7, 8, 9, 10, 11, 12, 13, 14, 15);
      acc = __builtin_amdgcn_wmma_f32_16x16x32_bf16(
          false, A, false, B, (short)0, acc, false, false);
    }

    // fold the replicated N columns: sum across each 16-lane half
#pragma unroll
    for (int v = 0; v < 8; ++v) {
      float s = acc[v];
      s += __shfl_xor(s, 1);
      s += __shfl_xor(s, 2);
      s += __shfl_xor(s, 4);
      s += __shfl_xor(s, 8);
      if ((lane & 15) == 0) {
        const int row = v + ((lane >> 4) << 3);      // lanes 0-15 -> rows 0-7, 16-31 -> 8-15
        atomicAdd(&z_lds[gw * 16 + row], s);         // ds_add_f32 (two k-halves combine)
      }
    }
    __syncthreads();

    // gate math + state update for this WG's 16 hidden units (wave 0, lanes 0-15)
    if (tid < 16) {
      const int r = tid;
      const int j = wgid * 16 + r;
      const float x  = sentence[t];
      const float zi = z_lds[r]      + wih_l[r]      * x + bias_l[r];
      const float zf = z_lds[16 + r] + wih_l[16 + r] * x + bias_l[16 + r];
      const float zg = z_lds[32 + r] + wih_l[32 + r] * x + bias_l[32 + r];
      const float zo = z_lds[48 + r] + wih_l[48 + r] * x + bias_l[48 + r];
      const float ig = sigf(zi), fg = sigf(zf), gg = tanhf(zg), og = sigf(zo);
      const float cn = fg * c_l[r] + ig * gg;
      c_l[r] = cn;
      const float hn = og * tanhf(cn);
      h_dst[j] = f2bf(hn);
      float p = wlin_l[r] * hn;                      // partial W_lin . h_new
      p += __shfl_xor(p, 1, 16);
      p += __shfl_xor(p, 2, 16);
      p += __shfl_xor(p, 4, 16);
      p += __shfl_xor(p, 8, 16);
      if (r == 0) atomicAdd(&out[t], p);             // out[t] pre-seeded with b_lin
    }
    __threadfence();
    __syncthreads();

    // grid-wide barrier: monotonic arrival counter (8192*64 < 2^32)
    if (tid == 0) {
      __hip_atomic_fetch_add(cnt, 1u, __ATOMIC_ACQ_REL, __HIP_MEMORY_SCOPE_AGENT);
      const unsigned target = (unsigned)(t + 1) * (unsigned)NWG;
      while (__hip_atomic_load(cnt, __ATOMIC_ACQUIRE, __HIP_MEMORY_SCOPE_AGENT) < target) {
        __builtin_amdgcn_s_sleep(1);
      }
    }
    __builtin_amdgcn_s_cluster_barrier();            // NOP when not clustered
    __syncthreads();
  }
}

extern "C" void kernel_launch(void* const* d_in, const int* in_sizes, int n_in,
                              void* d_out, int out_size, void* d_ws, size_t ws_size,
                              hipStream_t stream) {
  (void)in_sizes; (void)n_in; (void)out_size; (void)ws_size;
  const float* sentence = (const float*)d_in[0];
  const float* W_ih     = (const float*)d_in[1];
  const float* W_hh     = (const float*)d_in[2];
  const float* b_ih     = (const float*)d_in[3];
  const float* b_hh     = (const float*)d_in[4];
  const float* W_lin    = (const float*)d_in[5];
  const float* b_lin    = (const float*)d_in[6];
  float* out = (float*)d_out;

  unsigned*       cnt  = (unsigned*)d_ws;
  unsigned short* hbuf = (unsigned short*)((char*)d_ws + 128);

  lstm_init_kernel<<<(SEQ + TPB - 1) / TPB, TPB, 0, stream>>>(out, b_lin, cnt, hbuf);
  lstm_persistent_kernel<<<NWG, TPB, SMEM_BYTES, stream>>>(
      sentence, W_ih, W_hh, b_ih, b_hh, W_lin, out, cnt, hbuf);
}
